// NnBasic_77318001263141
// MI455X (gfx1250) — compile-verified, matrix operand order checked by reading, not run on hardware
//
#include <hip/hip_runtime.h>
#include <cmath>

// ---- CDNA5 vector types ----
typedef __attribute__((ext_vector_type(16))) _Float16 v16h;
typedef __attribute__((ext_vector_type(8)))  _Float16 v8h;
typedef __attribute__((ext_vector_type(4)))  _Float16 v4h;
typedef __attribute__((ext_vector_type(8)))  float    v8f;

#define FT_OUT 256
#define NACT   30
#define HDIM   512

__device__ __forceinline__ void add4(float4& a, const float4 b) {
  a.x += b.x; a.y += b.y; a.z += b.z; a.w += b.w;
}
__device__ __forceinline__ float clip01(float x) {
  return fminf(fmaxf(x, 0.0f), 1.0f);
}
__device__ __forceinline__ float4 hclip(const float4 a, const float4 b,
                                        const float4 c, const float4 d) {
  float4 r;
  r.x = clip01(a.x + b.x + c.x + d.x);
  r.y = clip01(a.y + b.y + c.y + d.y);
  r.z = clip01(a.z + b.z + c.z + d.z);
  r.w = clip01(a.w + b.w + c.w + d.w);
  return r;
}
__device__ __forceinline__ void store_h4(_Float16* p, const float4 v) {
  v4h t;
  t[0] = (_Float16)v.x; t[1] = (_Float16)v.y;
  t[2] = (_Float16)v.z; t[3] = (_Float16)v.w;
  *(v4h*)p = t;   // ds_store_b64
}

// One block = 16 boards (one WMMA M-tile) = 16 waves of 32.
// Phase 1: each wave gather-accumulates one board (f32, b128 row reads).
// Phase 2: wave 0 runs the 512->1 head as 16x v_wmma_f32_16x16x32_f16 with
//          out_w in column N=0 of the B fragment.
__global__ __launch_bounds__(512, 1) void nnue_fused(
    const int* __restrict__ stm_idx,  const int* __restrict__ nstm_idx,
    const int* __restrict__ f_stm_idx, const int* __restrict__ f_nstm_idx,
    const float* __restrict__ ft_w,  const float* __restrict__ ft_b,
    const float* __restrict__ fft_w, const float* __restrict__ fft_b,
    const float* __restrict__ out_w, const float* __restrict__ out_b,
    float* __restrict__ out, int n_boards)
{
  __shared__ _Float16 h16[16][HDIM];          // 16 KB of 320 KB/WGP LDS

  const int wave  = threadIdx.x >> 5;
  const int lane  = threadIdx.x & 31;
  const int board = blockIdx.x * 16 + wave;

  if (board < n_boards) {
    // --- lane-parallel index fetch, broadcast later via ds_bpermute ---
    int is = 0, in_ = 0, ifs = 0, ifn = 0;
    if (lane < NACT) {
      const int base = board * NACT + lane;
      is  = stm_idx[base];
      in_ = nstm_idx[base];
      ifs = f_stm_idx[base];
      ifn = f_nstm_idx[base];
    }

    float4 s0 = {0,0,0,0}, s1 = s0, n0 = s0, n1 = s0;
    float4 fs0 = s0, fs1 = s0, fn0 = s0, fn1 = s0;
    const float4* ftw  = reinterpret_cast<const float4*>(ft_w);
    const float4* fftw = reinterpret_cast<const float4*>(fft_w);

    for (int a = 0; a < NACT; ++a) {
      const int j1 = __shfl(is, a),  j2 = __shfl(in_, a);
      const int j3 = __shfl(ifs, a), j4 = __shfl(ifn, a);
      const float4* r1 = ftw  + (size_t)j1 * 64;   // 256 f32 = 64 float4/row
      const float4* r2 = ftw  + (size_t)j2 * 64;
      const float4* r3 = fftw + (size_t)j3 * 64;
      const float4* r4 = fftw + (size_t)j4 * 64;
      if (a + 1 < NACT) {                           // hide L2 latency on big table
        const int p1 = __shfl(is, a + 1), p2 = __shfl(in_, a + 1);
        __builtin_prefetch(ft_w + (size_t)p1 * 256 + lane * 8, 0, 0);
        __builtin_prefetch(ft_w + (size_t)p2 * 256 + lane * 8, 0, 0);
      }
      add4(s0,  r1[lane]); add4(s1,  r1[lane + 32]);   // global_load_b128 x8
      add4(n0,  r2[lane]); add4(n1,  r2[lane + 32]);
      add4(fs0, r3[lane]); add4(fs1, r3[lane + 32]);
      add4(fn0, r4[lane]); add4(fn1, r4[lane + 32]);
    }

    const float4 b0 = reinterpret_cast<const float4*>(ft_b)[lane];
    const float4 b1 = reinterpret_cast<const float4*>(ft_b)[lane + 32];
    const float4 g0 = reinterpret_cast<const float4*>(fft_b)[lane];
    const float4 g1 = reinterpret_cast<const float4*>(fft_b)[lane + 32];

    // merge + clip, stage to LDS as f16 rows (h in [0,1] -> <=5e-4 abs error)
    store_h4(&h16[wave][4 * lane],        hclip(s0, b0, fs0, g0));
    store_h4(&h16[wave][128 + 4 * lane],  hclip(s1, b1, fs1, g1));
    store_h4(&h16[wave][256 + 4 * lane],  hclip(n0, b0, fn0, g0));
    store_h4(&h16[wave][384 + 4 * lane],  hclip(n1, b1, fn1, g1));
  } else {
    for (int k = lane; k < HDIM; k += 32) h16[wave][k] = (_Float16)0.0f;
  }

  __syncthreads();

  // --- phase 2: 512->1 head on the matrix pipe (wave 0 only) ---
  if (wave == 0) {
    v8f acc = {0,0,0,0,0,0,0,0};
    const int m    = lane & 15;            // A row M / B column N
    const int koff = (lane >> 4) << 3;     // lanes 0-15 -> K 0..7/16..23; 16-31 -> 8..15/24..31
    const _Float16* hrow = &h16[m][0];
    const float sel = (m == 0) ? 1.0f : 0.0f;   // only B column N=0 carries out_w

#pragma unroll
    for (int kb = 0; kb < HDIM / 32; ++kb) {
      const int kbase = kb * 32;
      const v8h alo = *(const v8h*)(hrow + kbase + koff);        // ds_load_b128
      const v8h ahi = *(const v8h*)(hrow + kbase + 16 + koff);
      const v16h Amat = __builtin_shufflevector(
          alo, ahi, 0,1,2,3,4,5,6,7,8,9,10,11,12,13,14,15);
      v16h Bmat;
#pragma unroll
      for (int j = 0; j < 8; ++j) {
        Bmat[j]     = (_Float16)(sel * out_w[kbase + koff + j]);
        Bmat[8 + j] = (_Float16)(sel * out_w[kbase + 16 + koff + j]);
      }
      acc = __builtin_amdgcn_wmma_f32_16x16x32_f16(
          false, Amat, false, Bmat, (short)0, acc, false, false);
    }

    // D column N=0: lane 0 holds M=0..7 in acc[0..7], lane 16 holds M=8..15
    if (m == 0) {
      const float bias  = out_b[0];
      const int   mbase = (lane >> 4) * 8;
#pragma unroll
      for (int r = 0; r < 8; ++r) {
        const int b = blockIdx.x * 16 + mbase + r;
        if (b < n_boards) {
          const float z = acc[r] + bias;
          out[b] = 1.0f / (1.0f + expf(-z));
        }
      }
    }
  }
}

extern "C" void kernel_launch(void* const* d_in, const int* in_sizes, int n_in,
                              void* d_out, int out_size, void* d_ws, size_t ws_size,
                              hipStream_t stream) {
  const int*   stm   = (const int*)d_in[0];
  const int*   nstm  = (const int*)d_in[1];
  const int*   fstm  = (const int*)d_in[2];
  const int*   fnstm = (const int*)d_in[3];
  const float* ftw   = (const float*)d_in[4];
  const float* ftb   = (const float*)d_in[5];
  const float* fftw  = (const float*)d_in[6];
  const float* fftb  = (const float*)d_in[7];
  const float* outw  = (const float*)d_in[8];
  const float* outb  = (const float*)d_in[9];
  float* out = (float*)d_out;

  const int n_boards = in_sizes[0] / NACT;   // 16384
  dim3 grid((n_boards + 15) / 16);
  dim3 block(512);                           // 16 wave32 = one full WGP pass
  nnue_fused<<<grid, block, 0, stream>>>(stm, nstm, fstm, fnstm,
                                         ftw, ftb, fftw, fftb,
                                         outw, outb, out, n_boards);
}
